// dot_attention_74131135529648
// MI455X (gfx1250) — matrix operand order repaired and checked by reading, not compiled, not run on hardware
//
#include <hip/hip_runtime.h>
#include <cstdint>

// ---------------------------------------------------------------------------
// Fused scaled-dot-product attention forward for MI455X (gfx1250, wave32).
//   scores = Q K^T / sqrt(D); attn = softmax(scores); ctx = attn V
// Bandwidth-bound on the 256 MiB attn write (23.3 TB/s -> ~14us floor);
// matmuls use v_wmma_f32_16x16x32_bf16 (f32 accumulate).  Softmax row
// reductions use DPP16 butterflies (no LDS round-trips).  attn/ctx stores are
// non-temporal to keep the per-batch K/V working set resident in L2.
// ---------------------------------------------------------------------------

typedef __attribute__((ext_vector_type(16))) __bf16 v16bf;
typedef __attribute__((ext_vector_type(8)))  __bf16 v8bf;
typedef __attribute__((ext_vector_type(2)))  __bf16 v2bf;
typedef __attribute__((ext_vector_type(8)))  float  v8f;

constexpr int BATCH = 16;
constexpr int SEQ   = 2048;
constexpr int DIM   = 128;
constexpr int WAVES = 4;          // 4 x wave32 per workgroup
constexpr int KC    = 64;         // keys staged per chunk
constexpr float SCALE = 0.08838834764831845f;  // 1/sqrt(128)

// LDS row strides (bf16 elements), padded for conflict-free b128 reads.
constexpr int QB_S = DIM + 8;     // 136 el = 272 B (17 * 16B)
constexpr int KB_S = DIM + 8;     // 136 el
constexpr int VT_S = KC  + 8;     // 72 el = 144 B (9 * 16B)
constexpr int PB_S = 32  + 8;     // 40 el = 80 B (5 * 16B)

__device__ inline __bf16 f2bf(float x) {            // scalar RNE convert
  uint32_t u = __builtin_bit_cast(uint32_t, x);
  u += 0x7FFFu + ((u >> 16) & 1u);
  return __builtin_bit_cast(__bf16, (uint16_t)(u >> 16));
}

__device__ inline v2bf pk2(float a, float b) {
  v2bf r; r[0] = f2bf(a); r[1] = f2bf(b); return r;
}
__device__ inline uint2 pk4(float4 f) {             // 4 f32 -> 4 bf16 (8 B)
  uint2 u;
  u.x = __builtin_bit_cast(uint32_t, pk2(f.x, f.y));
  u.y = __builtin_bit_cast(uint32_t, pk2(f.z, f.w));
  return u;
}

__device__ inline v16bf cat16(v8bf lo, v8bf hi) {
  v16bf r;
#pragma unroll
  for (int i = 0; i < 8; ++i) { r[i] = lo[i]; r[i + 8] = hi[i]; }
  return r;
}

// ---- 16-lane-group butterfly reductions via DPP16 (VALU only) ----
#if __has_builtin(__builtin_amdgcn_update_dpp)
template <int CTRL>
__device__ inline float dppmov(float x) {
  int r = __builtin_amdgcn_update_dpp(0, __builtin_bit_cast(int, x),
                                      CTRL, 0xF, 0xF, true);
  return __builtin_bit_cast(float, r);
}
__device__ inline float rmax16(float x) {
  x = fmaxf(x, dppmov<0x0B1>(x));   // quad_perm:[1,0,3,2]  (xor 1)
  x = fmaxf(x, dppmov<0x04E>(x));   // quad_perm:[2,3,0,1]  (xor 2)
  x = fmaxf(x, dppmov<0x141>(x));   // row_half_mirror      (xor 4-equiv)
  x = fmaxf(x, dppmov<0x140>(x));   // row_mirror           (xor 8-equiv)
  return x;
}
__device__ inline float rsum16(float x) {
  x += dppmov<0x0B1>(x);
  x += dppmov<0x04E>(x);
  x += dppmov<0x141>(x);
  x += dppmov<0x140>(x);
  return x;
}
#else
__device__ inline float rmax16(float x) {
#pragma unroll
  for (int off = 1; off < 16; off <<= 1) x = fmaxf(x, __shfl_xor(x, off, 32));
  return x;
}
__device__ inline float rsum16(float x) {
#pragma unroll
  for (int off = 1; off < 16; off <<= 1) x += __shfl_xor(x, off, 32);
  return x;
}
#endif

__global__ __launch_bounds__(WAVES * 32)
void attn_fwd_wmma(const float* __restrict__ q,
                   const float* __restrict__ k,
                   const float* __restrict__ v,
                   float* __restrict__ ctx,
                   float* __restrict__ attn) {
  __shared__ __bf16 Qb[WAVES * 16 * QB_S];          // Q rows, row-major bf16
  __shared__ __bf16 Kb[KC * KB_S];                  // K chunk, row-major bf16
  __shared__ __bf16 Vt[DIM * VT_S];                 // V chunk, TRANSPOSED [d][k]
  __shared__ __bf16 Pb[WAVES * 16 * PB_S];          // per-wave prob tile 16x32

  const int tid  = threadIdx.x;
  const int wave = tid >> 5;
  const int lane = tid & 31;
  const int g    = lane >> 4;                       // lane group (0/1)
  const int ln   = lane & 15;

  const int tile  = blockIdx.x * WAVES + wave;      // one 16-row q-tile / wave
  const int batch = tile >> 7;                      // 128 tiles per batch
  const int qt    = tile & 127;
  const int qrow0_blk = ((blockIdx.x * WAVES) & 127) * 16;  // block's 64 q rows

  const size_t bQ = ((size_t)batch * SEQ + qrow0_blk) * DIM;
  const size_t bK = (size_t)batch * SEQ * DIM;
  constexpr int NCHUNK = SEQ / KC;                  // 32

  // ---- stage Q (64 rows x 128 cols f32 -> bf16 LDS), coalesced float4 ----
#pragma unroll
  for (int i = 0; i < 16; ++i) {
    int idx = tid + i * 128;                        // float4 index
    int row = idx >> 5;                             // 32 float4 per row
    int c   = (idx & 31) * 4;
    float4 f = *(const float4*)(q + bQ + (size_t)row * DIM + c);
    *(uint2*)&Qb[row * QB_S + c] = pk4(f);          // ds_store_b64
  }
  __syncthreads();

  // ---- load this wave's Q A-fragments (16x32 bf16 each, 4 d-chunks) ----
  v16bf qf[4];
#pragma unroll
  for (int dc = 0; dc < 4; ++dc) {
    const __bf16* p = &Qb[(wave * 16 + ln) * QB_S + dc * 32 + g * 16];
    qf[dc] = cat16(*(const v8bf*)p, *(const v8bf*)(p + 8));
  }

  float m[8], l[8];
#pragma unroll
  for (int r = 0; r < 8; ++r) { m[r] = -3.0e38f; l[r] = 0.0f; }

  // ======================= PASS 1: row max / sumexp ======================
  for (int ck = 0; ck < NCHUNK; ++ck) {
    if (ck + 1 < NCHUNK) {                          // global_prefetch_b8
      const char* nk = (const char*)(k + bK + (size_t)((ck + 1) * KC) * DIM);
      __builtin_prefetch(nk + tid * 256, 0, 3);     // 128 thr * 256B = 32 KiB
    }
    __syncthreads();                                // Kb free to overwrite
#pragma unroll
    for (int i = 0; i < 16; ++i) {                  // stage K chunk (bf16)
      int idx = tid + i * 128;
      int row = idx >> 5, c = (idx & 31) * 4;
      float4 f = *(const float4*)(k + bK + (size_t)(ck * KC + row) * DIM + c);
      *(uint2*)&Kb[row * KB_S + c] = pk4(f);
    }
    __syncthreads();

#pragma unroll
    for (int kt = 0; kt < 4; ++kt) {                // 4 key tiles of 16
      v16bf kb[4];                                  // batch all B-frag loads
#pragma unroll
      for (int dc = 0; dc < 4; ++dc) {
        const __bf16* p = &Kb[(kt * 16 + ln) * KB_S + dc * 32 + g * 16];
        kb[dc] = cat16(*(const v8bf*)p, *(const v8bf*)(p + 8));
      }
      v8f c8;
#pragma unroll
      for (int i = 0; i < 8; ++i) c8[i] = 0.0f;
#pragma unroll
      for (int dc = 0; dc < 4; ++dc)
        c8 = __builtin_amdgcn_wmma_f32_16x16x32_bf16(
                 false, qf[dc], false, kb[dc], (short)0, c8, false, false);
#pragma unroll
      for (int r = 0; r < 8; ++r) {                 // online softmax stats
        float s  = c8[r] * SCALE;
        float mn = fmaxf(m[r], rmax16(s));
        float ps = rsum16(__expf(s - mn));
        l[r] = l[r] * __expf(m[r] - mn) + ps;
        m[r] = mn;
      }
    }
  }

  float invl[8];
#pragma unroll
  for (int r = 0; r < 8; ++r) invl[r] = 1.0f / l[r];

  v8f acc[8];                                       // ctx accumulators 16x128
#pragma unroll
  for (int dt = 0; dt < 8; ++dt)
#pragma unroll
    for (int i = 0; i < 8; ++i) acc[dt][i] = 0.0f;

  // ============== PASS 2: attn write + ctx = P V (WMMA) ==================
  for (int ck = 0; ck < NCHUNK; ++ck) {
    if (ck + 1 < NCHUNK) {
      size_t noff = bK + (size_t)((ck + 1) * KC) * DIM;
      __builtin_prefetch((const char*)(k + noff) + tid * 256, 0, 3);
      __builtin_prefetch((const char*)(v + noff) + tid * 256, 0, 3);
    }
    __syncthreads();
#pragma unroll
    for (int i = 0; i < 16; ++i) {                  // stage K + V^T chunks
      int idx = tid + i * 128;
      int row = idx >> 5, c = (idx & 31) * 4;
      size_t go = bK + (size_t)(ck * KC + row) * DIM + c;
      float4 fk = *(const float4*)(k + go);
      *(uint2*)&Kb[row * KB_S + c] = pk4(fk);
      float4 fv = *(const float4*)(v + go);
      v2bf v01 = pk2(fv.x, fv.y), v23 = pk2(fv.z, fv.w);
      Vt[(c + 0) * VT_S + row] = v01[0];            // transpose into [d][k]
      Vt[(c + 1) * VT_S + row] = v01[1];
      Vt[(c + 2) * VT_S + row] = v23[0];
      Vt[(c + 3) * VT_S + row] = v23[1];
    }
    __syncthreads();

#pragma unroll
    for (int sub = 0; sub < 2; ++sub) {             // 32-key sub-chunks
#pragma unroll
      for (int t = 0; t < 2; ++t) {
        int kt = sub * 2 + t;
        v16bf kb[4];
#pragma unroll
        for (int dc = 0; dc < 4; ++dc) {
          const __bf16* p = &Kb[(kt * 16 + ln) * KB_S + dc * 32 + g * 16];
          kb[dc] = cat16(*(const v8bf*)p, *(const v8bf*)(p + 8));
        }
        v8f c8;
#pragma unroll
        for (int i = 0; i < 8; ++i) c8[i] = 0.0f;
#pragma unroll
        for (int dc = 0; dc < 4; ++dc)
          c8 = __builtin_amdgcn_wmma_f32_16x16x32_bf16(
                   false, qf[dc], false, kb[dc], (short)0, c8, false, false);
#pragma unroll
        for (int r = 0; r < 8; ++r) {
          float pr = __expf(c8[r] * SCALE - m[r]) * invl[r];
          size_t ai = ((size_t)batch * SEQ + (qt * 16 + r + 8 * g)) * (size_t)SEQ
                    + (size_t)(ck * KC + kt * 16 + ln);
          __builtin_nontemporal_store(pr, attn + ai);   // stream, keep L2 for K/V
          Pb[(wave * 16 + r + 8 * g) * PB_S + t * 16 + ln] = f2bf(pr);
        }
      }
      __syncthreads();                              // Pb visible wave-wide

      const __bf16* pp = &Pb[(wave * 16 + ln) * PB_S + g * 16];
      v16bf pa = cat16(*(const v8bf*)pp, *(const v8bf*)(pp + 8));

      // ctx += P * V, 2-deep software pipeline on the V B-fragments
      v16bf vb[2];
      {
        const __bf16* p0 = &Vt[(ln) * VT_S + sub * 32 + g * 16];
        vb[0] = cat16(*(const v8bf*)p0, *(const v8bf*)(p0 + 8));
      }
#pragma unroll
      for (int dt = 0; dt < 8; ++dt) {
        if (dt < 7) {
          const __bf16* p = &Vt[((dt + 1) * 16 + ln) * VT_S + sub * 32 + g * 16];
          vb[(dt + 1) & 1] = cat16(*(const v8bf*)p, *(const v8bf*)(p + 8));
        }
        acc[dt] = __builtin_amdgcn_wmma_f32_16x16x32_bf16(
                      false, pa, false, vb[dt & 1], (short)0, acc[dt], false, false);
      }
      __syncthreads();                              // Pb reads done before reuse
    }
  }

  // ---- epilogue: write context f32 (non-temporal, write-once) ----
#pragma unroll
  for (int dt = 0; dt < 8; ++dt)
#pragma unroll
    for (int r = 0; r < 8; ++r) {
      size_t ci = ((size_t)batch * SEQ + (qt * 16 + r + 8 * g)) * DIM
                + (size_t)(dt * 16 + ln);
      __builtin_nontemporal_store(acc[dt][r], ctx + ci);
    }
}

extern "C" void kernel_launch(void* const* d_in, const int* in_sizes, int n_in,
                              void* d_out, int out_size, void* d_ws, size_t ws_size,
                              hipStream_t stream) {
  (void)in_sizes; (void)n_in; (void)out_size; (void)d_ws; (void)ws_size;
  const float* q = (const float*)d_in[0];
  const float* k = (const float*)d_in[1];
  const float* v = (const float*)d_in[2];
  float* ctx  = (float*)d_out;                                  // [B,S,D]
  float* attn = (float*)d_out + (size_t)BATCH * SEQ * DIM;      // [B,S,S]

  dim3 grid((BATCH * (SEQ / 16)) / WAVES);   // 512 blocks
  dim3 block(WAVES * 32);                    // 4 wave32
  attn_fwd_wmma<<<grid, block, 0, stream>>>(q, k, v, ctx, attn);
}
